// GatedRNN_a_24515673325783
// MI455X (gfx1250) — matrix operand order repaired and checked by reading, not compiled, not run on hardware
//
#include <hip/hip_runtime.h>
#include <hip/hip_bf16.h>

#define B_  512
#define T_  2048
#define NI_ 24
#define NH_ 64
#define NO_ 24

typedef __attribute__((ext_vector_type(16))) _Float16 v16h;
typedef __attribute__((ext_vector_type(8)))  _Float16 v8h;
typedef __attribute__((ext_vector_type(8)))  float    v8f;

__device__ __forceinline__ float fexp(float x) {
  return __builtin_amdgcn_exp2f(x * 1.44269504088896340736f);
}
__device__ __forceinline__ float fsigmoid(float x) {
  return __builtin_amdgcn_rcpf(1.0f + fexp(-x));
}
__device__ __forceinline__ float ftanh(float x) {
  float e = fexp(2.0f * x);                 // large +x -> inf -> rcp 0 -> 1 ; large -x -> 0 -> -1
  return 1.0f - 2.0f * __builtin_amdgcn_rcpf(e + 1.0f);
}

__device__ __forceinline__ v8f wmma(v16h a, v16h b, v8f c) {
  return __builtin_amdgcn_wmma_f32_16x16x32_f16(false, a, false, b, (short)0, c, false, false);
}

// B-operand fragment (32x16 f16, K x N) from LDS stored as [Npad][Kpad].
// Lane layout: N = lane&15 (+16*ntile); halves 0-7 = K = kbase+(lane>=16?8:0)+0..7, halves 8-15 = +16.
__device__ __forceinline__ v16h load_b(const _Float16* s, int Kpad, int ntile, int kbase, int lane) {
  const _Float16* p = s + ((ntile * 16) + (lane & 15)) * Kpad + kbase + ((lane >> 4) * 8);
  v8h lo = *(const v8h*)p;
  v8h hi = *(const v8h*)(p + 16);
  v16h r;
#pragma unroll
  for (int i = 0; i < 8; ++i) { r[i] = lo[i]; r[i + 8] = hi[i]; }
  return r;
}

// A-operand fragment (16x32 f16, M x K) from LDS stored as [16][Kpad].
// Lane layout: M = lane&15; halves 0-7 = K = kbase+(lane>=16?8:0)+0..7, halves 8-15 = +16.
__device__ __forceinline__ v16h load_a(const _Float16* s, int Kpad, int kbase, int lane) {
  const _Float16* p = s + (lane & 15) * Kpad + kbase + ((lane >> 4) * 8);
  v8h lo = *(const v8h*)p;
  v8h hi = *(const v8h*)(p + 16);
  v16h r;
#pragma unroll
  for (int i = 0; i < 8; ++i) { r[i] = lo[i]; r[i + 8] = hi[i]; }
  return r;
}

// Store one 16x16 C tile (f32 accumulator layout: vgpr j -> M=j+(lane>=16?8:0), N=lane&15)
// into an A-layout f16 staging buffer [16][Kpad] at column block ntile*16.
__device__ __forceinline__ void store_cfrag(_Float16* s, int Kpad, int ntile, v8f v, int lane) {
  int hi = lane >> 4, nl = lane & 15;
#pragma unroll
  for (int j = 0; j < 8; ++j)
    s[(j + hi * 8) * Kpad + ntile * 16 + nl] = (_Float16)v[j];
}

// Copy weight W[N][K] (f32, row-major) -> LDS [Npad][Kpad] f16 with zero padding.
__device__ void load_weight(const float* __restrict__ g, _Float16* s,
                            int N, int K, int Npad, int Kpad, int tid) {
  for (int i = tid; i < Npad * Kpad; i += 32) {
    int n = i / Kpad, k = i - n * Kpad;
    float v = (n < N && k < K) ? g[n * K + k] : 0.0f;
    s[i] = (_Float16)v;
  }
}

__global__ __launch_bounds__(32) void gated_rnn_kernel(
    const float* __restrict__ X,
    const float* __restrict__ Wih1, const float* __restrict__ Whh1,
    const float* __restrict__ bih1, const float* __restrict__ bhh1,
    const float* __restrict__ Wih2, const float* __restrict__ Whh2,
    const float* __restrict__ bih2, const float* __restrict__ bhh2,
    const float* __restrict__ Wo1,  const float* __restrict__ bo1,
    const float* __restrict__ Wo2,  const float* __restrict__ bo2,
    const float* __restrict__ Wg,   const float* __restrict__ bg,
    const float* __restrict__ W1,   const float* __restrict__ b1,
    const float* __restrict__ W2,   const float* __restrict__ b2,
    float* __restrict__ Out)
{
  // Weights resident in LDS as f16, B-operand friendly [N][Kpad] (59392 B)
  __shared__ __align__(32) _Float16 sWih1[64 * 32];   // K 24->32
  __shared__ __align__(32) _Float16 sWhh1[64 * 64];
  __shared__ __align__(32) _Float16 sWih2[64 * 64];
  __shared__ __align__(32) _Float16 sWhh2[64 * 64];
  __shared__ __align__(32) _Float16 sWo1 [32 * 64];   // N 24->32
  __shared__ __align__(32) _Float16 sWo2 [32 * 64];
  __shared__ __align__(32) _Float16 sWg  [64 * 128];
  __shared__ __align__(32) _Float16 sW1  [32 * 64];
  __shared__ __align__(32) _Float16 sW2  [32 * 32];
  // A-layout staging (5120 B)
  __shared__ __align__(32) _Float16 sC [16 * 32];
  __shared__ __align__(32) _Float16 sH1[16 * 64];
  __shared__ __align__(32) _Float16 sH2[16 * 64];

  const int lane = threadIdx.x & 31;
  const int hi = lane >> 4, nl = lane & 15;
  const int m0 = blockIdx.x * 16;

  load_weight(Wih1, sWih1, 64, 24, 64, 32, lane);
  load_weight(Whh1, sWhh1, 64, 64, 64, 64, lane);
  load_weight(Wih2, sWih2, 64, 64, 64, 64, lane);
  load_weight(Whh2, sWhh2, 64, 64, 64, 64, lane);
  load_weight(Wo1,  sWo1,  24, 64, 32, 64, lane);
  load_weight(Wo2,  sWo2,  24, 64, 32, 64, lane);
  load_weight(Wg,   sWg,   64, 128, 64, 128, lane);
  load_weight(W1,   sW1,   24, 64, 32, 64, lane);
  load_weight(W2,   sW2,   24, 24, 32, 32, lane);
  for (int i = lane; i < 16 * 64; i += 32) { sH1[i] = (_Float16)0.f; sH2[i] = (_Float16)0.f; }
  __syncthreads();

  // per-lane biases (lane's column n = nt*16 + nl)
  float bb1[4], bb2[4], bgv[4], bo1v[2], bo2v[2], b1v[2], b2v[2];
#pragma unroll
  for (int nt = 0; nt < 4; ++nt) {
    int n = nt * 16 + nl;
    bb1[nt] = bih1[n] + bhh1[n];
    bb2[nt] = bih2[n] + bhh2[n];
    bgv[nt] = bg[n];
  }
#pragma unroll
  for (int nt = 0; nt < 2; ++nt) {
    int n = nt * 16 + nl;
    bo1v[nt] = (n < NO_) ? bo1[n] : 0.f;
    bo2v[nt] = (n < NO_) ? bo2[n] : 0.f;
    b1v[nt]  = (n < NI_) ? b1[n]  : 0.f;
    b2v[nt]  = (n < NI_) ? b2[n]  : 0.f;
  }

  // persistent state in C layout (f32 registers)
  v8f h2c[4], uc[4], a1c[2], a2c[2];
#pragma unroll
  for (int nt = 0; nt < 4; ++nt) {
    h2c[nt] = {};
    v8f one; 
#pragma unroll
    for (int j = 0; j < 8; ++j) one[j] = 1.0f;
    uc[nt] = one;
  }
#pragma unroll
  for (int nt = 0; nt < 2; ++nt) {
    a2c[nt] = {};
    v8f one;
#pragma unroll
    for (int j = 0; j < 8; ++j) one[j] = 1.0f;
    a1c[nt] = one;
  }

  for (int t = 0; t < T_; ++t) {
    // ---- c = a1*x1 + a2*x2 (C layout), stage to sC in A layout (K padded to 32)
#pragma unroll
    for (int nt = 0; nt < 2; ++nt) {
      int f = nt * 16 + nl;
#pragma unroll
      for (int j = 0; j < 8; ++j) {
        int m = m0 + j + hi * 8;
        float cv = 0.f;
        if (f < NI_) {
          const float* xr = X + ((size_t)m * T_ + t) * (2 * NI_);
          cv = a1c[nt][j] * xr[f] + a2c[nt][j] * xr[NI_ + f];
        }
        sC[(j + hi * 8) * 32 + f] = (_Float16)cv;
      }
    }
    __syncthreads();

    // ---- hid1 = tanh(c@Wih1^T + hid1@Whh1^T + b)
    v16h aC   = load_a(sC,  32, 0,  lane);
    v16h aH1a = load_a(sH1, 64, 0,  lane);
    v16h aH1b = load_a(sH1, 64, 32, lane);
    v8f h1c[4];
#pragma unroll
    for (int nt = 0; nt < 4; ++nt) {
      v8f acc = {};
      acc = wmma(aC,   load_b(sWih1, 32, nt, 0,  lane), acc);
      acc = wmma(aH1a, load_b(sWhh1, 64, nt, 0,  lane), acc);
      acc = wmma(aH1b, load_b(sWhh1, 64, nt, 32, lane), acc);
#pragma unroll
      for (int j = 0; j < 8; ++j) h1c[nt][j] = ftanh(acc[j] + bb1[nt]);
    }
    __syncthreads();
#pragma unroll
    for (int nt = 0; nt < 4; ++nt) store_cfrag(sH1, 64, nt, h1c[nt], lane);
    __syncthreads();

    // ---- cand2 = tanh(hid1@Wih2^T + hid2@Whh2^T + b); hid2 = u*cand2 + (1-u)*hid2
    v16h aH1na = load_a(sH1, 64, 0,  lane);
    v16h aH1nb = load_a(sH1, 64, 32, lane);
    v16h aH2a  = load_a(sH2, 64, 0,  lane);
    v16h aH2b  = load_a(sH2, 64, 32, lane);
#pragma unroll
    for (int nt = 0; nt < 4; ++nt) {
      v8f acc = {};
      acc = wmma(aH1na, load_b(sWih2, 64, nt, 0,  lane), acc);
      acc = wmma(aH1nb, load_b(sWih2, 64, nt, 32, lane), acc);
      acc = wmma(aH2a,  load_b(sWhh2, 64, nt, 0,  lane), acc);
      acc = wmma(aH2b,  load_b(sWhh2, 64, nt, 32, lane), acc);
#pragma unroll
      for (int j = 0; j < 8; ++j) {
        float cand = ftanh(acc[j] + bb2[nt]);
        h2c[nt][j] = uc[nt][j] * cand + (1.0f - uc[nt][j]) * h2c[nt][j];
      }
    }
    __syncthreads();
#pragma unroll
    for (int nt = 0; nt < 4; ++nt) store_cfrag(sH2, 64, nt, h2c[nt], lane);
    __syncthreads();
    v16h aH2na = load_a(sH2, 64, 0,  lane);
    v16h aH2nb = load_a(sH2, 64, 32, lane);

    // ---- out = tanh(hid1@Wo1^T + bo1) + tanh(hid2@Wo2^T + bo2)
#pragma unroll
    for (int nt = 0; nt < 2; ++nt) {
      v8f acc1 = {}, acc2 = {};
      acc1 = wmma(aH1na, load_b(sWo1, 64, nt, 0,  lane), acc1);
      acc1 = wmma(aH1nb, load_b(sWo1, 64, nt, 32, lane), acc1);
      acc2 = wmma(aH2na, load_b(sWo2, 64, nt, 0,  lane), acc2);
      acc2 = wmma(aH2nb, load_b(sWo2, 64, nt, 32, lane), acc2);
      int n = nt * 16 + nl;
      if (n < NO_) {
#pragma unroll
        for (int j = 0; j < 8; ++j) {
          float ov = ftanh(acc1[j] + bo1v[nt]) + ftanh(acc2[j] + bo2v[nt]);
          int m = m0 + j + hi * 8;
          Out[((size_t)m * T_ + t) * NO_ + n] = ov;
        }
      }
    }

    // ---- u = sigmoid([hid1,hid2]@Wg^T + bg)   (K = 128)
#pragma unroll
    for (int nt = 0; nt < 4; ++nt) {
      v8f acc = {};
      acc = wmma(aH1na, load_b(sWg, 128, nt, 0,  lane), acc);
      acc = wmma(aH1nb, load_b(sWg, 128, nt, 32, lane), acc);
      acc = wmma(aH2na, load_b(sWg, 128, nt, 64, lane), acc);
      acc = wmma(aH2nb, load_b(sWg, 128, nt, 96, lane), acc);
#pragma unroll
      for (int j = 0; j < 8; ++j) uc[nt][j] = fsigmoid(acc[j] + bgv[nt]);
    }

    // ---- t = hid2@W1^T + b1 ; e_i = exp(tanh(t + x_i@W2^T + b2)); a_i = e_i/(e1+e2)
    {
      // x1/x2 A fragments direct from global (rows contiguous: 48 floats per (m,t))
      const float* xr = X + ((size_t)(m0 + nl) * T_ + t) * (2 * NI_);
      int k0 = hi * 8;
      v16h x1a, x2a;
#pragma unroll
      for (int i = 0; i < 8; ++i) {
        x1a[i]     = (_Float16)xr[k0 + i];            // K = 0..15  (valid, <24)
        x2a[i]     = (_Float16)xr[NI_ + k0 + i];
        x1a[i + 8] = hi ? (_Float16)0.f : (_Float16)xr[16 + i];        // K = 16..23, 24..31 -> 0
        x2a[i + 8] = hi ? (_Float16)0.f : (_Float16)xr[NI_ + 16 + i];
      }
#pragma unroll
      for (int nt = 0; nt < 2; ++nt) {
        v8f accT = {};
        accT = wmma(aH2na, load_b(sW1, 64, nt, 0,  lane), accT);
        accT = wmma(aH2nb, load_b(sW1, 64, nt, 32, lane), accT);
        v16h bw2 = load_b(sW2, 32, nt, 0, lane);
        v8f acc1 = {}, acc2 = {};
        acc1 = wmma(x1a, bw2, acc1);
        acc2 = wmma(x2a, bw2, acc2);
#pragma unroll
        for (int j = 0; j < 8; ++j) {
          float tv = accT[j] + b1v[nt];
          float e1 = fexp(ftanh(tv + acc1[j] + b2v[nt]));
          float e2 = fexp(ftanh(tv + acc2[j] + b2v[nt]));
          float r  = __builtin_amdgcn_rcpf(e1 + e2);
          a1c[nt][j] = e1 * r;
          a2c[nt][j] = e2 * r;
        }
      }
    }
  }
}

extern "C" void kernel_launch(void* const* d_in, const int* in_sizes, int n_in,
                              void* d_out, int out_size, void* d_ws, size_t ws_size,
                              hipStream_t stream) {
  const float* X    = (const float*)d_in[0];
  const float* Wih1 = (const float*)d_in[1];
  const float* Whh1 = (const float*)d_in[2];
  const float* bih1 = (const float*)d_in[3];
  const float* bhh1 = (const float*)d_in[4];
  const float* Wih2 = (const float*)d_in[5];
  const float* Whh2 = (const float*)d_in[6];
  const float* bih2 = (const float*)d_in[7];
  const float* bhh2 = (const float*)d_in[8];
  const float* Wo1  = (const float*)d_in[9];
  const float* bo1  = (const float*)d_in[10];
  const float* Wo2  = (const float*)d_in[11];
  const float* bo2  = (const float*)d_in[12];
  const float* Wg   = (const float*)d_in[13];
  const float* bg   = (const float*)d_in[14];
  const float* W1   = (const float*)d_in[15];
  const float* b1   = (const float*)d_in[16];
  const float* W2   = (const float*)d_in[17];
  const float* b2   = (const float*)d_in[18];
  float* Out = (float*)d_out;

  dim3 grid(B_ / 16), block(32);
  gated_rnn_kernel<<<grid, block, 0, stream>>>(
      X, Wih1, Whh1, bih1, bhh1, Wih2, Whh2, bih2, bhh2,
      Wo1, bo1, Wo2, bo2, Wg, bg, W1, b1, W2, b2, Out);
}
